// MoRGPT_32358283608141
// MI455X (gfx1250) — compile-verified, matrix operand order checked by reading, not run on hardware
//
#include <hip/hip_runtime.h>
#include <hip/hip_bf16.h>

typedef _Float16 f16;
typedef __attribute__((ext_vector_type(16))) _Float16 v16h;
typedef __attribute__((ext_vector_type(8)))  float    v8f;

#define WMMA16(a, b, c) \
  __builtin_amdgcn_wmma_f32_16x16x32_f16(false, (a), false, (b), (short)0, (c), false, false)

// ---------------- model constants ----------------
#define Vv   32000
#define Cc   768
#define Hh   12
#define Tt   1024
#define Bb   2
#define Dd   4
#define HIDd 384
#define FFf  3072
#define HDd  64
#define Mrows (Bb * Tt)   // 2048

// k-index within a 16x32 f16 fragment, CDNA5 layout (half-lane K split)
__device__ __forceinline__ int frag_k(int j, int half) {
  return ((j < 4) ? (2 * j) : (16 + 2 * (j - 4))) + half * 8;
}

// ---------------- elementwise kernels ----------------
__global__ void k_f32_to_f16(const float* __restrict__ in, f16* __restrict__ out, size_t n) {
  size_t i = (size_t)blockIdx.x * blockDim.x + threadIdx.x;
  if (i < n) out[i] = (f16)in[i];
}

// transpose convert: in [K,N] f32 row-major -> out [N,K] f16 row-major
__global__ void k_f32_to_f16_t(const float* __restrict__ in, f16* __restrict__ out,
                               int K_, int N_) {
  size_t e = (size_t)blockIdx.x * blockDim.x + threadIdx.x;
  if (e >= (size_t)K_ * N_) return;
  int k = (int)(e % K_);
  int n = (int)(e / K_);
  out[e] = (f16)in[(size_t)k * N_ + n];
}

__global__ void k_embed(const int* __restrict__ idx, const float* __restrict__ tok,
                        const float* __restrict__ pos, float* __restrict__ x32,
                        f16* __restrict__ x16) {
  size_t e = (size_t)blockIdx.x * blockDim.x + threadIdx.x;
  if (e >= (size_t)Mrows * Cc) return;
  int c = (int)(e % Cc);
  size_t bt = e / Cc;
  int t = (int)(bt % Tt);
  float v = tok[(size_t)idx[bt] * Cc + c] + pos[(size_t)t * Cc + c];
  x32[e] = v;
  x16[e] = (f16)v;
}

// q,k -> [B,H,T,HD]; v -> [B,H,HD,T] (transposed so attention can vector-stage Vt)
__global__ void k_qkv_split(const float* __restrict__ qkv, f16* __restrict__ q,
                            f16* __restrict__ k, f16* __restrict__ v) {
  size_t e = (size_t)blockIdx.x * blockDim.x + threadIdx.x;
  if (e >= (size_t)Mrows * 3 * Cc) return;
  int c3 = (int)(e % (3 * Cc));
  size_t bt = e / (3 * Cc);
  int t = (int)(bt % Tt);
  int b = (int)(bt / Tt);
  int which = c3 / Cc;
  int cc = c3 % Cc;
  int hh = cc / HDd;
  int d = cc % HDd;
  f16 val = (f16)qkv[e];
  if (which == 0)
    q[(((size_t)b * Hh + hh) * Tt + t) * HDd + d] = val;
  else if (which == 1)
    k[(((size_t)b * Hh + hh) * Tt + t) * HDd + d] = val;
  else
    v[(((size_t)b * Hh + hh) * HDd + d) * Tt + t] = val;
}

__global__ void k_accum(const float* __restrict__ h, const int* __restrict__ sel,
                        int d, float* __restrict__ acc) {
  size_t e = (size_t)blockIdx.x * blockDim.x + threadIdx.x;
  if (e >= (size_t)Mrows * Cc) return;
  if (sel[e / Cc] == d) acc[e] += h[e];
}

__global__ void k_router2(const float* __restrict__ rh, const float* __restrict__ w2,
                          const float* __restrict__ b2, int* __restrict__ sel) {
  int row = blockIdx.x * blockDim.x + threadIdx.x;
  if (row >= Mrows) return;
  float best = -3.4e38f;
  int bi = 0;
  const float* r = rh + (size_t)row * HIDd;
#pragma unroll
  for (int d = 0; d < Dd; ++d) {
    float acc = b2[d];
    for (int k = 0; k < HIDd; ++k) acc += r[k] * w2[k * Dd + d];
    if (acc > best) { best = acc; bi = d; }
  }
  sel[row] = bi;
}

// ---------------- LayerNorm (f32 in, f16 out) ----------------
__global__ __launch_bounds__(256) void k_layernorm(const float* __restrict__ x,
                                                   const float* __restrict__ gs,
                                                   const float* __restrict__ gb,
                                                   f16* __restrict__ y) {
  __shared__ float red[256];
  int row = blockIdx.x;
  const float* xr = x + (size_t)row * Cc;
  int tid = threadIdx.x;
  float s = 0.f;
  for (int c = tid; c < Cc; c += 256) s += xr[c];
  red[tid] = s;
  __syncthreads();
  for (int o = 128; o > 0; o >>= 1) {
    if (tid < o) red[tid] += red[tid + o];
    __syncthreads();
  }
  float mean = red[0] / (float)Cc;
  __syncthreads();
  float vv = 0.f;
  for (int c = tid; c < Cc; c += 256) {
    float d = xr[c] - mean;
    vv += d * d;
  }
  red[tid] = vv;
  __syncthreads();
  for (int o = 128; o > 0; o >>= 1) {
    if (tid < o) red[tid] += red[tid + o];
    __syncthreads();
  }
  float inv = rsqrtf(red[0] / (float)Cc + 1e-5f);
  for (int c = tid; c < Cc; c += 256)
    y[(size_t)row * Cc + c] = (f16)((xr[c] - mean) * inv * gs[c] + gb[c]);
}

// ---------------- WMMA GEMM: C[M,N] = act(A[M,K] x B + bias) + resid ----------------
// A: f16 row-major [M,K]. Bt: f16 [N,K] (pre-transposed weights -> staging is pure
// vector copy, rows contiguous in k). act: 0 none, 1 relu, 2 exact gelu.
// Block tile 128x128, 8 waves of 32x64. LDS fragment-major As[m][k], Bs[n][k]
// => all WMMA operand loads are two contiguous 8-half runs per lane (ds_load_b128).
__global__ __launch_bounds__(256) void k_gemm_f16(const f16* __restrict__ A,
                                                  const f16* __restrict__ Bt,
                                                  const float* __restrict__ bias,
                                                  const float* __restrict__ resid,
                                                  float* __restrict__ Cout,
                                                  int M_, int N_, int K_, int act) {
  __shared__ __align__(16) f16 As[128][40];   // [m][k]
  __shared__ __align__(16) f16 Bs[128][40];   // [n][k]
  int tid = threadIdx.x;
  int wave = tid >> 5, lane = tid & 31;
  int wm = wave >> 1, wn = wave & 1;
  int m0 = blockIdx.y * 128, n0 = blockIdx.x * 128;
  int mlo = lane & 15, half = lane >> 4;

  v8f acc[2][4];
#pragma unroll
  for (int i = 0; i < 2; ++i)
#pragma unroll
    for (int j = 0; j < 4; ++j)
#pragma unroll
      for (int r = 0; r < 8; ++r) acc[i][j][r] = 0.f;

  for (int k0 = 0; k0 < K_; k0 += 32) {
    // stage A tile 128x32 and B tile 128x32 (both vector copies)
#pragma unroll
    for (int it = 0; it < 2; ++it) {
      int s = tid + it * 256;
      int r = s >> 2, c8 = (s & 3) << 3;
      *(uint4*)(&As[r][c8]) =
          *(const uint4*)(A + (size_t)(m0 + r) * K_ + k0 + c8);
      *(uint4*)(&Bs[r][c8]) =
          *(const uint4*)(Bt + (size_t)(n0 + r) * K_ + k0 + c8);
    }
    if (k0 + 32 < K_) {
      __builtin_prefetch(A + (size_t)(m0 + (tid >> 1)) * K_ + k0 + 32, 0, 1);
      __builtin_prefetch(Bt + (size_t)(n0 + (tid >> 1)) * K_ + k0 + 32, 0, 1);
    }
    __syncthreads();

    v16h af[2];
#pragma unroll
    for (int sm = 0; sm < 2; ++sm) {
      const f16* Ar = &As[wm * 32 + sm * 16 + mlo][0];
#pragma unroll
      for (int j = 0; j < 8; ++j) {
        int kb = frag_k(j, half);
        af[sm][2 * j] = Ar[kb];
        af[sm][2 * j + 1] = Ar[kb + 1];
      }
    }
#pragma unroll
    for (int sn = 0; sn < 4; ++sn) {
      v16h bf;
      const f16* Br = &Bs[wn * 64 + sn * 16 + mlo][0];
#pragma unroll
      for (int j = 0; j < 8; ++j) {
        int kb = frag_k(j, half);
        bf[2 * j] = Br[kb];
        bf[2 * j + 1] = Br[kb + 1];
      }
      acc[0][sn] = WMMA16(af[0], bf, acc[0][sn]);
      acc[1][sn] = WMMA16(af[1], bf, acc[1][sn]);
    }
    __syncthreads();
  }

#pragma unroll
  for (int sm = 0; sm < 2; ++sm)
#pragma unroll
    for (int sn = 0; sn < 4; ++sn) {
      int rbase = m0 + wm * 32 + sm * 16 + half * 8;
      int col = n0 + wn * 64 + sn * 16 + mlo;
#pragma unroll
      for (int r = 0; r < 8; ++r) {
        float v = acc[sm][sn][r];
        if (bias) v += bias[col];
        if (act == 1) v = fmaxf(v, 0.f);
        else if (act == 2) v = 0.5f * v * (1.f + erff(v * 0.70710678f));
        size_t off = (size_t)(rbase + r) * N_ + col;
        if (resid) v += resid[off];
        Cout[off] = v;
      }
    }
}

// ---------------- WMMA flash attention ----------------
// Q,K: f16 [B,H,T,64]. V: f16 [B,H,64,T] (pre-transposed). O: f32 [B,T,C].
// One block = 128 q rows of one (b,h). Ks [key][hd] and Vt [hd][kv] are both
// fragment-contiguous, staged with pure vector copies.
__global__ __launch_bounds__(256) void k_attention(const f16* __restrict__ Q,
                                                   const f16* __restrict__ K,
                                                   const f16* __restrict__ V,
                                                   float* __restrict__ O) {
  const int QT = Tt / 128;  // 8
  int qt = blockIdx.x % QT;
  int h = (blockIdx.x / QT) % Hh;
  int b = blockIdx.x / (QT * Hh);
  size_t base = ((size_t)(b * Hh + h)) * Tt * HDd;  // same size for both layouts

  int tid = threadIdx.x;
  int wave = tid >> 5, lane = tid & 31;
  int mlo = lane & 15, half = lane >> 4;
  int qbase = qt * 128 + wave * 16;

  __shared__ __align__(16) f16 Ks[128][72];     // [key][hd]
  __shared__ __align__(16) f16 Vt[64][136];     // [hd][kv]
  __shared__ __align__(16) f16 Ps[8][16][136];  // per-wave P, [m][kv]

  // Q fragments (16 rows x 64 hd = 2 k-steps), loaded straight from global
  v16h qf[2];
  {
    const f16* Qr = Q + base + (size_t)(qbase + mlo) * HDd;
#pragma unroll
    for (int s = 0; s < 2; ++s)
#pragma unroll
      for (int j = 0; j < 8; ++j) {
        int kb = frag_k(j, half) + s * 32;
        qf[s][2 * j] = Qr[kb];
        qf[s][2 * j + 1] = Qr[kb + 1];
      }
  }

  v8f of[4];
  float rmax[8], rsum[8];
#pragma unroll
  for (int j = 0; j < 4; ++j)
#pragma unroll
    for (int r = 0; r < 8; ++r) of[j][r] = 0.f;
#pragma unroll
  for (int r = 0; r < 8; ++r) { rmax[r] = -INFINITY; rsum[r] = 0.f; }

  const float scale = 0.125f;  // 1/sqrt(64)

  for (int kt = 0; kt <= qt; ++kt) {
    // stage K tile [128 key][64 hd] (vector copy)
#pragma unroll
    for (int it = 0; it < 4; ++it) {
      int s = tid + it * 256;
      int r = s >> 3, c8 = (s & 7) << 3;
      *(uint4*)(&Ks[r][c8]) =
          *(const uint4*)(K + base + (size_t)(kt * 128 + r) * HDd + c8);
    }
    // stage V tile [64 hd][128 kv] (vector copy from transposed global layout)
#pragma unroll
    for (int it = 0; it < 4; ++it) {
      int s = tid + it * 256;
      int r = s >> 4, c8 = (s & 15) << 3;
      *(uint4*)(&Vt[r][c8]) =
          *(const uint4*)(V + base + (size_t)r * Tt + kt * 128 + c8);
    }
    __syncthreads();

    // S = Q K^T  (16 q-rows x 128 keys per wave)
    v8f sf[8];
#pragma unroll
    for (int nt = 0; nt < 8; ++nt) {
      v8f c;
#pragma unroll
      for (int r = 0; r < 8; ++r) c[r] = 0.f;
#pragma unroll
      for (int s2 = 0; s2 < 2; ++s2) {
        v16h bf;
        const f16* Kr = &Ks[nt * 16 + mlo][0];
#pragma unroll
        for (int j = 0; j < 8; ++j) {
          int kb = frag_k(j, half) + s2 * 32;
          bf[2 * j] = Kr[kb];
          bf[2 * j + 1] = Kr[kb + 1];
        }
        c = WMMA16(qf[s2], bf, c);
      }
      sf[nt] = c;
    }

    // scale + causal mask on diagonal tile
#pragma unroll
    for (int nt = 0; nt < 8; ++nt)
#pragma unroll
      for (int r = 0; r < 8; ++r) {
        float v = sf[nt][r] * scale;
        if (kt == qt) {
          int col = nt * 16 + mlo;
          int row = wave * 16 + r + 8 * half;
          if (col > row) v = -INFINITY;
        }
        sf[nt][r] = v;
      }

    // online softmax
    float fo[8];
#pragma unroll
    for (int r = 0; r < 8; ++r) {
      float m = sf[0][r];
#pragma unroll
      for (int nt = 1; nt < 8; ++nt) m = fmaxf(m, sf[nt][r]);
      for (int d2 = 1; d2 < 16; d2 <<= 1) m = fmaxf(m, __shfl_xor(m, d2, 32));
      float nm = fmaxf(rmax[r], m);
      fo[r] = expf(rmax[r] - nm);
      rmax[r] = nm;
    }
    float psum[8];
#pragma unroll
    for (int r = 0; r < 8; ++r) psum[r] = 0.f;
#pragma unroll
    for (int nt = 0; nt < 8; ++nt)
#pragma unroll
      for (int r = 0; r < 8; ++r) {
        float p = expf(sf[nt][r] - rmax[r]);
        sf[nt][r] = p;
        psum[r] += p;
      }
#pragma unroll
    for (int r = 0; r < 8; ++r) {
      for (int d2 = 1; d2 < 16; d2 <<= 1) psum[r] += __shfl_xor(psum[r], d2, 32);
      rsum[r] = rsum[r] * fo[r] + psum[r];
    }
    // rescale running O
#pragma unroll
    for (int j = 0; j < 4; ++j)
#pragma unroll
      for (int r = 0; r < 8; ++r) of[j][r] *= fo[r];

    // write P (C-layout) to LDS for transpose into A-layout
#pragma unroll
    for (int nt = 0; nt < 8; ++nt)
#pragma unroll
      for (int r = 0; r < 8; ++r)
        Ps[wave][r + 8 * half][nt * 16 + mlo] = (f16)sf[nt][r];
    __syncthreads();

    // O += P V   (contraction over 128 kv = 4 WMMA steps)
#pragma unroll
    for (int kk = 0; kk < 4; ++kk) {
      v16h aP;
      const f16* Pr = &Ps[wave][mlo][0];
#pragma unroll
      for (int j = 0; j < 8; ++j) {
        int kb = frag_k(j, half) + kk * 32;
        aP[2 * j] = Pr[kb];
        aP[2 * j + 1] = Pr[kb + 1];
      }
#pragma unroll
      for (int jhd = 0; jhd < 4; ++jhd) {
        v16h bV;
        const f16* Vr = &Vt[jhd * 16 + mlo][0];
#pragma unroll
        for (int j = 0; j < 8; ++j) {
          int kb = frag_k(j, half) + kk * 32;
          bV[2 * j] = Vr[kb];
          bV[2 * j + 1] = Vr[kb + 1];
        }
        of[jhd] = WMMA16(aP, bV, of[jhd]);
      }
    }
    __syncthreads();
  }

  // normalize + store to [B,T,C]
#pragma unroll
  for (int jhd = 0; jhd < 4; ++jhd)
#pragma unroll
    for (int r = 0; r < 8; ++r) {
      int row = qt * 128 + wave * 16 + r + 8 * half;
      float v = of[jhd][r] / rsum[r];
      O[((size_t)b * Tt + row) * Cc + h * HDd + jhd * 16 + mlo] = v;
    }
}

// ---------------- host side ----------------
static inline void* carve(char*& p, size_t bytes) {
  void* r = (void*)p;
  p += (bytes + 255) & ~(size_t)255;
  return r;
}

extern "C" void kernel_launch(void* const* d_in, const int* in_sizes, int n_in,
                              void* d_out, int out_size, void* d_ws, size_t ws_size,
                              hipStream_t stream) {
  const int* idx = (const int*)d_in[0];
  const float* tok_emb = (const float*)d_in[1];
  const float* pos_emb = (const float*)d_in[2];
  const float* r_w1 = (const float*)d_in[3];
  const float* r_b1 = (const float*)d_in[4];
  const float* r_w2 = (const float*)d_in[5];
  const float* r_b2 = (const float*)d_in[6];
  const float* ln1_s = (const float*)d_in[7];
  const float* ln1_b = (const float*)d_in[8];
  const float* qkv_w = (const float*)d_in[9];
  const float* proj_w = (const float*)d_in[10];
  const float* proj_b = (const float*)d_in[11];
  const float* ln2_s = (const float*)d_in[12];
  const float* ln2_b = (const float*)d_in[13];
  const float* f_w1 = (const float*)d_in[14];
  const float* f_b1 = (const float*)d_in[15];
  const float* f_w2 = (const float*)d_in[16];
  const float* f_b2 = (const float*)d_in[17];
  const float* lnf_s = (const float*)d_in[18];
  const float* lnf_b = (const float*)d_in[19];
  const float* head_w = (const float*)d_in[20];
  float* logits = (float*)d_out;

  char* p = (char*)d_ws;
  // weights pre-transposed to [N,K] f16
  f16* qkvw16 = (f16*)carve(p, (size_t)Cc * 3 * Cc * 2);
  f16* projw16 = (f16*)carve(p, (size_t)Cc * Cc * 2);
  f16* fw116 = (f16*)carve(p, (size_t)Cc * FFf * 2);
  f16* fw216 = (f16*)carve(p, (size_t)FFf * Cc * 2);
  f16* headw16 = (f16*)carve(p, (size_t)Vv * Cc * 2);  // already [N,K]
  f16* rw116 = (f16*)carve(p, (size_t)Cc * HIDd * 2);
  float* h32 = (float*)carve(p, (size_t)Mrows * Cc * 4);
  f16* x16 = (f16*)carve(p, (size_t)Mrows * Cc * 2);
  f16* y16 = (f16*)carve(p, (size_t)Mrows * Cc * 2);
  float* z32 = (float*)carve(p, (size_t)Mrows * Cc * 4);
  float* acc32 = (float*)carve(p, (size_t)Mrows * Cc * 4);
  float* rh32 = (float*)carve(p, (size_t)Mrows * HIDd * 4);
  int* sel = (int*)carve(p, (size_t)Mrows * 4);
  float* qkv32 = (float*)carve(p, (size_t)Mrows * 3 * Cc * 4);
  f16* q16 = (f16*)carve(p, (size_t)Mrows * Cc * 2);
  f16* k16 = (f16*)carve(p, (size_t)Mrows * Cc * 2);
  f16* v16 = (f16*)carve(p, (size_t)Mrows * Cc * 2);
  float* o32 = (float*)carve(p, (size_t)Mrows * Cc * 4);
  f16* o16 = (f16*)carve(p, (size_t)Mrows * Cc * 2);
  float* ff32 = (float*)carve(p, (size_t)Mrows * FFf * 4);
  f16* ff16 = (f16*)carve(p, (size_t)Mrows * FFf * 2);
  f16* xf16 = (f16*)carve(p, (size_t)Mrows * Cc * 2);
  (void)ws_size; (void)n_in; (void)in_sizes; (void)out_size;

  auto conv = [&](const float* src, f16* dst, size_t n) {
    k_f32_to_f16<<<(unsigned)((n + 255) / 256), 256, 0, stream>>>(src, dst, n);
  };
  auto convT = [&](const float* src, f16* dst, int K_, int N_) {
    size_t n = (size_t)K_ * N_;
    k_f32_to_f16_t<<<(unsigned)((n + 255) / 256), 256, 0, stream>>>(src, dst, K_, N_);
  };
  convT(qkv_w, qkvw16, Cc, 3 * Cc);   // -> [3C, C]
  convT(proj_w, projw16, Cc, Cc);     // -> [C, C]^T
  convT(f_w1, fw116, Cc, FFf);        // -> [FF, C]
  convT(f_w2, fw216, FFf, Cc);        // -> [C, FF]
  convT(r_w1, rw116, Cc, HIDd);       // -> [HID, C]
  conv(head_w, headw16, (size_t)Vv * Cc);  // head_w already [V, C] = [N, K]

  hipMemsetAsync(acc32, 0, (size_t)Mrows * Cc * 4, stream);

  // embeddings -> h32 (f32 working state) and x16 (router input)
  {
    size_t n = (size_t)Mrows * Cc;
    k_embed<<<(unsigned)((n + 255) / 256), 256, 0, stream>>>(idx, tok_emb, pos_emb, h32, x16);
  }

  auto gemm = [&](const f16* A, const f16* Bt, const float* bias, const float* resid,
                  float* Cout, int M_, int N_, int K_, int act) {
    dim3 grid(N_ / 128, M_ / 128);
    k_gemm_f16<<<grid, 256, 0, stream>>>(A, Bt, bias, resid, Cout, M_, N_, K_, act);
  };

  // router: rh = relu(x @ r_w1 + b1); logits+argmax
  gemm(x16, rw116, r_b1, nullptr, rh32, Mrows, HIDd, Cc, 1);
  k_router2<<<Mrows / 256, 256, 0, stream>>>(rh32, r_w2, r_b2, sel);

  for (int d = 0; d < Dd; ++d) {
    k_layernorm<<<Mrows, 256, 0, stream>>>(h32, ln1_s, ln1_b, y16);
    gemm(y16, qkvw16, nullptr, nullptr, qkv32, Mrows, 3 * Cc, Cc, 0);
    {
      size_t n = (size_t)Mrows * 3 * Cc;
      k_qkv_split<<<(unsigned)((n + 255) / 256), 256, 0, stream>>>(qkv32, q16, k16, v16);
    }
    k_attention<<<Bb * Hh * (Tt / 128), 256, 0, stream>>>(q16, k16, v16, o32);
    conv(o32, o16, (size_t)Mrows * Cc);
    gemm(o16, projw16, proj_b, h32, z32, Mrows, Cc, Cc, 0);
    k_layernorm<<<Mrows, 256, 0, stream>>>(z32, ln2_s, ln2_b, y16);
    gemm(y16, fw116, f_b1, nullptr, ff32, Mrows, FFf, Cc, 2);
    conv(ff32, ff16, (size_t)Mrows * FFf);
    gemm(ff16, fw216, f_b2, z32, h32, Mrows, Cc, FFf, 0);
    {
      size_t n = (size_t)Mrows * Cc;
      k_accum<<<(unsigned)((n + 255) / 256), 256, 0, stream>>>(h32, sel, d, acc32);
    }
  }

  k_layernorm<<<Mrows, 256, 0, stream>>>(acc32, lnf_s, lnf_b, xf16);
  // head: logits = xf @ head_w^T
  gemm(xf16, headw16, nullptr, nullptr, logits, Mrows, Vv, Cc, 0);
}
